// MultiHeadLayer_24223615549948
// MI455X (gfx1250) — compile-verified, matrix operand order checked by reading, not run on hardware
//
#include <hip/hip_runtime.h>

// MHA fused kernel for MI455X (gfx1250, wave32, WMMA bf16, pipelined async-LDS)
// B=1024, S=128, E=256, H=8, D=64, HD=512, SCALING=False, use_res=True, relu

#define BB 1024
#define SS 128
#define EE 256
#define HH 8
#define DD 64
#define HD 512

typedef __attribute__((ext_vector_type(16))) __bf16 v16bf;
typedef __attribute__((ext_vector_type(8)))  __bf16 v8bf;
typedef __attribute__((ext_vector_type(4)))  __bf16 v4bf;
typedef __attribute__((ext_vector_type(8)))  float  v8f;

// 16-byte payload type matching the async-LDS builtin's expected pointee
typedef int b128_t __attribute__((vector_size(16)));

#if defined(__has_builtin)
#if __has_builtin(__builtin_amdgcn_global_load_async_to_lds_b128)
#define HAS_ASYNC_LDS 1
#endif
#endif
#ifndef HAS_ASYNC_LDS
#define HAS_ASYNC_LDS 0
#endif

// Copy 16 bytes global -> LDS (async on gfx1250)
__device__ __forceinline__ void copy16_g2l(const __bf16* gsrc, __bf16* ldst) {
#if HAS_ASYNC_LDS
    __builtin_amdgcn_global_load_async_to_lds_b128(
        (__attribute__((address_space(1))) b128_t*)(void*)gsrc,
        (__attribute__((address_space(3))) b128_t*)(void*)ldst, 0, 0);
#else
    *(uint4*)ldst = *(const uint4*)gsrc;
#endif
}

#if HAS_ASYNC_LDS
#define ASYNC_WAIT_6() asm volatile("s_wait_asynccnt 0x6" ::: "memory")
#define ASYNC_WAIT_0() asm volatile("s_wait_asynccnt 0x0" ::: "memory")
#else
#define ASYNC_WAIT_6() ((void)0)
#define ASYNC_WAIT_0() ((void)0)
#endif

// ---------------- Shared tail: writeback + attention + epilogue -------------
__device__ __forceinline__ void mha_tail(
    __bf16* smem, v8f (&acc)[4][4],
    int b, int h, int w, int lh, int hi, int mat, int half,
    float* __restrict__ Out)
{
    __bf16* Qb = smem;            // [128][64]
    __bf16* Kb = smem + 8192;     // [128][64]
    __bf16* Vt = smem + 16384;    // [64][128] transposed
    __bf16* Rb = smem + 24576;    // [128][64]
    __bf16* Pb = smem;            // [128][128] aliases Qb+Kb

    const v8f vzero = {0.f,0.f,0.f,0.f,0.f,0.f,0.f,0.f};

    // Write projection results to LDS (V transposed; R as bf16 residual)
    #pragma unroll
    for (int mt = 0; mt < 4; ++mt) {
        #pragma unroll
        for (int nt = 0; nt < 4; ++nt) {
            int col  = nt * 16 + lh;
            int row0 = half * 64 + mt * 16 + hi * 8;
            #pragma unroll
            for (int r = 0; r < 8; ++r) {
                int row = row0 + r;
                __bf16 v = (__bf16)acc[mt][nt][r];
                if      (mat == 0) Qb[row * 64 + col]  = v;
                else if (mat == 1) Kb[row * 64 + col]  = v;
                else if (mat == 2) Vt[col * 128 + row] = v;   // transposed
                else               Rb[row * 64 + col]  = v;
            }
        }
    }
    __syncthreads();

    // Phase 2: scores = Q K^T; wave w owns rows [16w,16w+16) -> local softmax
    v8f sc[8];
    #pragma unroll
    for (int nt = 0; nt < 8; ++nt) sc[nt] = vzero;

    #pragma unroll
    for (int kc = 0; kc < 2; ++kc) {
        v16bf aq = *(const v16bf*)&Qb[(w * 16 + lh) * 64 + kc * 32 + hi * 16];
        #pragma unroll
        for (int nt = 0; nt < 8; ++nt) {
            v16bf bk = *(const v16bf*)&Kb[(nt * 16 + lh) * 64 + kc * 32 + hi * 16];
            sc[nt] = __builtin_amdgcn_wmma_f32_16x16x32_bf16(
                false, aq, false, bk, (short)0, sc[nt], false, false);
        }
    }

    #pragma unroll
    for (int r = 0; r < 8; ++r) {
        float m = sc[0][r];
        #pragma unroll
        for (int nt = 1; nt < 8; ++nt) m = fmaxf(m, sc[nt][r]);
        #pragma unroll
        for (int off = 8; off >= 1; off >>= 1)
            m = fmaxf(m, __shfl_xor(m, off, 32));
        float sum = 0.f;
        #pragma unroll
        for (int nt = 0; nt < 8; ++nt) {
            float e = __expf(sc[nt][r] - m);
            sc[nt][r] = e;
            sum += e;
        }
        #pragma unroll
        for (int off = 8; off >= 1; off >>= 1)
            sum += __shfl_xor(sum, off, 32);
        float inv = 1.0f / sum;
        #pragma unroll
        for (int nt = 0; nt < 8; ++nt) sc[nt][r] *= inv;
    }

    __syncthreads();   // done reading Qb/Kb before P overwrites them

    #pragma unroll
    for (int nt = 0; nt < 8; ++nt) {
        int col  = nt * 16 + lh;
        int row0 = w * 16 + hi * 8;
        #pragma unroll
        for (int r = 0; r < 8; ++r)
            Pb[(row0 + r) * 128 + col] = (__bf16)sc[nt][r];
    }
    __syncthreads();

    // Phase 3: out = P @ V
    v8f o[4];
    #pragma unroll
    for (int nt = 0; nt < 4; ++nt) o[nt] = vzero;

    #pragma unroll
    for (int kc = 0; kc < 4; ++kc) {
        v16bf ap = *(const v16bf*)&Pb[(w * 16 + lh) * 128 + kc * 32 + hi * 16];
        #pragma unroll
        for (int nt = 0; nt < 4; ++nt) {
            v16bf bv = *(const v16bf*)&Vt[(nt * 16 + lh) * 128 + kc * 32 + hi * 16];
            o[nt] = __builtin_amdgcn_wmma_f32_16x16x32_bf16(
                false, ap, false, bv, (short)0, o[nt], false, false);
        }
    }

    // Phase 4: + residual, ReLU, store
    float* Ob = Out + (size_t)b * SS * HD + (size_t)h * 64;
    #pragma unroll
    for (int nt = 0; nt < 4; ++nt) {
        int col  = nt * 16 + lh;
        int row0 = w * 16 + hi * 8;
        #pragma unroll
        for (int r = 0; r < 8; ++r) {
            int row = row0 + r;
            float v = o[nt][r] + (float)Rb[row * 64 + col];
            Ob[(size_t)row * HD + col] = fmaxf(v, 0.0f);
        }
    }
}

// ---------------- Pre-pass kernels: fp32 -> bf16 mirrors in workspace -------
__global__ __launch_bounds__(256) void convert_x_kernel(
    const float* __restrict__ X, __bf16* __restrict__ Xbf)
{
    int i = blockIdx.x * blockDim.x + threadIdx.x;   // one float4 per thread
    float4 v = ((const float4*)X)[i];
    v4bf o;
    o[0] = (__bf16)v.x; o[1] = (__bf16)v.y; o[2] = (__bf16)v.z; o[3] = (__bf16)v.w;
    *(v4bf*)&Xbf[(size_t)i * 4] = o;
}

__global__ __launch_bounds__(256) void convert_w_kernel(
    const float* __restrict__ Wq, const float* __restrict__ Wk,
    const float* __restrict__ Wv, const float* __restrict__ Wr,
    __bf16* __restrict__ WtAll)   // [4][HD][EE] = W^T per matrix, bf16
{
    int t  = blockIdx.x * blockDim.x + threadIdx.x;  // 65536 threads
    int m  = t >> 14;            // 0..3 (block-uniform)
    int n  = (t >> 5) & 511;     // output column (HD)
    int k0 = (t & 31) * 8;       // 8 consecutive k (EE)
    const float* Wp = (m == 0) ? Wq : (m == 1) ? Wk : (m == 2) ? Wv : Wr;
    v8bf o;
    #pragma unroll
    for (int j = 0; j < 8; ++j)
        o[j] = (__bf16)Wp[(k0 + j) * HD + n];
    *(v8bf*)&WtAll[((size_t)(m * 512 + n)) * EE + k0] = o;
}

// Stage one k-chunk (X tile + 4 transposed weight tiles) into an LDS buffer.
__device__ __forceinline__ void stage_chunk(
    const __bf16* __restrict__ Xb, const __bf16* __restrict__ WtAll,
    int h, int tid, int k0, __bf16* buf)
{
    __bf16* Xc = buf;           // [128][32]
    __bf16* Wt = buf + 4096;    // [4][64][32]
    #pragma unroll
    for (int i = 0; i < 2; ++i) {
        int idx = tid + i * 256;          // 0..511
        int row = idx >> 2, seg = idx & 3;
        copy16_g2l(&Xb[row * EE + k0 + seg * 8], &Xc[row * 32 + seg * 8]);
    }
    #pragma unroll
    for (int j = 0; j < 4; ++j) {
        int idx = tid + j * 256;          // 0..1023
        int m   = idx >> 8;
        int n   = (idx >> 2) & 63;
        int seg = idx & 3;
        copy16_g2l(&WtAll[((size_t)(m * 512 + h * 64 + n)) * EE + k0 + seg * 8],
                   &Wt[(m * 64 + n) * 32 + seg * 8]);
    }
}

// ---------------- Main fused kernel (bf16, double-buffered async staging) ---
__global__ __launch_bounds__(256, 1) void mha_fused_bf16_kernel(
    const __bf16* __restrict__ Xbf,    // [B][S][E] bf16
    const __bf16* __restrict__ WtAll,  // [4][HD][EE] bf16 (transposed)
    float* __restrict__ Out)           // [B][S][HD]
{
    __shared__ __align__(32) __bf16 smem[32768];
    // Staging double buffer: elems [0,12288) and [12288,24576) -- both alias
    // the Q/K/V output regions, which are written only after the loop.

    const int b    = blockIdx.x >> 3;
    const int h    = blockIdx.x & 7;
    const int tid  = threadIdx.x;
    const int w    = tid >> 5;
    const int lane = tid & 31;
    const int lh   = lane & 15;
    const int hi   = lane >> 4;
    const int mat  = w >> 1;
    const int half = w & 1;

    const __bf16* Xb = Xbf + (size_t)b * SS * EE;

    const v8f vzero = {0.f,0.f,0.f,0.f,0.f,0.f,0.f,0.f};
    v8f acc[4][4];
    #pragma unroll
    for (int i = 0; i < 4; ++i)
        #pragma unroll
        for (int j = 0; j < 4; ++j)
            acc[i][j] = vzero;

    // Prologue: stage chunk 0 into buffer 0
    stage_chunk(Xb, WtAll, h, tid, 0, smem);

    for (int kc = 0; kc < EE / 32; ++kc) {
        __bf16* cur = smem + (kc & 1) * 12288;

        if (kc < EE / 32 - 1) {
            // Issue next chunk into the other buffer, then drain only the
            // older 6 per-wave async ops (in-order completion => chunk kc done)
            stage_chunk(Xb, WtAll, h, tid, (kc + 1) * 32, smem + ((kc + 1) & 1) * 12288);
            ASYNC_WAIT_6();
        } else {
            ASYNC_WAIT_0();
        }
        __syncthreads();   // all waves' fills for chunk kc visible

        __bf16* Xc = cur;
        __bf16* Wt = cur + 4096;

        v16bf a[4];
        #pragma unroll
        for (int mt = 0; mt < 4; ++mt) {
            int row = half * 64 + mt * 16 + lh;
            a[mt] = *(const v16bf*)&Xc[row * 32 + hi * 16];
        }
        #pragma unroll
        for (int nt = 0; nt < 4; ++nt) {
            v16bf bt = *(const v16bf*)&Wt[(mat * 64 + nt * 16 + lh) * 32 + hi * 16];
            #pragma unroll
            for (int mt = 0; mt < 4; ++mt)
                acc[mt][nt] = __builtin_amdgcn_wmma_f32_16x16x32_bf16(
                    false, a[mt], false, bt, (short)0, acc[mt][nt], false, false);
        }
        __syncthreads();   // reads done before this buffer is refilled
    }

    mha_tail(smem, acc, b, h, w, lh, hi, mat, half, Out);
}

// ---------------- Fallback fused kernel (fp32 inputs, sync staging) ---------
__global__ __launch_bounds__(256, 1) void mha_fused_kernel(
    const float* __restrict__ X,
    const float* __restrict__ Wq, const float* __restrict__ Wk,
    const float* __restrict__ Wv, const float* __restrict__ Wr,
    float* __restrict__ Out)
{
    __shared__ __align__(32) __bf16 smem[32768];
    __bf16* Xc = smem;
    __bf16* Wt = smem + 4096;

    const int b    = blockIdx.x >> 3;
    const int h    = blockIdx.x & 7;
    const int tid  = threadIdx.x;
    const int w    = tid >> 5;
    const int lane = tid & 31;
    const int lh   = lane & 15;
    const int hi   = lane >> 4;
    const int mat  = w >> 1;
    const int half = w & 1;

    const float* Xb = X + (size_t)b * SS * EE;
    const v8f vzero = {0.f,0.f,0.f,0.f,0.f,0.f,0.f,0.f};
    v8f acc[4][4];
    #pragma unroll
    for (int i = 0; i < 4; ++i)
        #pragma unroll
        for (int j = 0; j < 4; ++j)
            acc[i][j] = vzero;

    const int xrow = tid >> 1;
    const int xcb  = (tid & 1) * 16;
    const int wk   = tid >> 3;
    const int wnh  = tid & 7;

    for (int kc = 0; kc < EE / 32; ++kc) {
        const int k0 = kc * 32;
        {
            const float4* xp = (const float4*)&Xb[xrow * EE + k0 + xcb];
            float4 x0 = xp[0], x1 = xp[1], x2 = xp[2], x3 = xp[3];
            v16bf xv;
            xv[0]=(__bf16)x0.x;  xv[1]=(__bf16)x0.y;  xv[2]=(__bf16)x0.z;  xv[3]=(__bf16)x0.w;
            xv[4]=(__bf16)x1.x;  xv[5]=(__bf16)x1.y;  xv[6]=(__bf16)x1.z;  xv[7]=(__bf16)x1.w;
            xv[8]=(__bf16)x2.x;  xv[9]=(__bf16)x2.y;  xv[10]=(__bf16)x2.z; xv[11]=(__bf16)x2.w;
            xv[12]=(__bf16)x3.x; xv[13]=(__bf16)x3.y; xv[14]=(__bf16)x3.z; xv[15]=(__bf16)x3.w;
            *(v16bf*)&Xc[xrow * 32 + xcb] = xv;
        }
        #pragma unroll
        for (int m = 0; m < 4; ++m) {
            const float* Wp = (m == 0) ? Wq : (m == 1) ? Wk : (m == 2) ? Wv : Wr;
            const float4* wp = (const float4*)&Wp[(k0 + wk) * HD + h * 64 + wnh * 8];
            float4 a0 = wp[0], a1 = wp[1];
            float vals[8] = {a0.x, a0.y, a0.z, a0.w, a1.x, a1.y, a1.z, a1.w};
            #pragma unroll
            for (int jj = 0; jj < 8; ++jj)
                Wt[(m * 64 + wnh * 8 + jj) * 32 + wk] = (__bf16)vals[jj];
        }
        __syncthreads();

        v16bf a[4];
        #pragma unroll
        for (int mt = 0; mt < 4; ++mt) {
            int row = half * 64 + mt * 16 + lh;
            a[mt] = *(const v16bf*)&Xc[row * 32 + hi * 16];
        }
        #pragma unroll
        for (int nt = 0; nt < 4; ++nt) {
            v16bf bt = *(const v16bf*)&Wt[(mat * 64 + nt * 16 + lh) * 32 + hi * 16];
            #pragma unroll
            for (int mt = 0; mt < 4; ++mt)
                acc[mt][nt] = __builtin_amdgcn_wmma_f32_16x16x32_bf16(
                    false, a[mt], false, bt, (short)0, acc[mt][nt], false, false);
        }
        __syncthreads();
    }

    mha_tail(smem, acc, b, h, w, lh, hi, mat, half, Out);
}

extern "C" void kernel_launch(void* const* d_in, const int* in_sizes, int n_in,
                              void* d_out, int out_size, void* d_ws, size_t ws_size,
                              hipStream_t stream) {
    const float* X  = (const float*)d_in[0];
    const float* Wq = (const float*)d_in[1];
    const float* Wk = (const float*)d_in[2];
    const float* Wv = (const float*)d_in[3];
    const float* Wr = (const float*)d_in[4];
    float* Out = (float*)d_out;

    const size_t xElems = (size_t)BB * SS * EE;          // 33,554,432
    const size_t wElems = (size_t)4 * HD * EE;           // 524,288
    const size_t need   = (xElems + wElems) * sizeof(__bf16);

    dim3 block(256);
    if (ws_size >= need) {
        __bf16* Xbf   = (__bf16*)d_ws;
        __bf16* WtAll = Xbf + xElems;
        convert_x_kernel<<<dim3((unsigned)(xElems / 4 / 256)), block, 0, stream>>>(X, Xbf);
        convert_w_kernel<<<dim3(256), block, 0, stream>>>(Wq, Wk, Wv, Wr, WtAll);
        mha_fused_bf16_kernel<<<dim3(BB * HH), block, 0, stream>>>(Xbf, WtAll, Out);
    } else {
        mha_fused_kernel<<<dim3(BB * HH), block, 0, stream>>>(X, Wq, Wk, Wv, Wr, Out);
    }
}
